// BiLSTMGCNBehaviorModel_34059090657792
// MI455X (gfx1250) — compile-verified, matrix operand order checked by reading, not run on hardware
//
#include <hip/hip_runtime.h>
#include <hip/hip_bf16.h>

typedef unsigned short u16;
typedef unsigned int   u32;
typedef unsigned long long u64;

typedef __attribute__((ext_vector_type(16))) __bf16 v16bf;
typedef __attribute__((ext_vector_type(8)))  float  v8f;
typedef __attribute__((ext_vector_type(4)))  unsigned int v4u;
typedef __attribute__((ext_vector_type(8)))  int v8i;
typedef __attribute__((ext_vector_type(4)))  int v4i;

union V16 { v16bf v; u32 u[8]; };

#define USE_TDM 1

__device__ __forceinline__ u16 f2bf(float f) {
    u32 u = __float_as_uint(f);
    u32 r = (u + 0x7FFFu + ((u >> 16) & 1u)) >> 16;
    return (u16)r;
}
__device__ __forceinline__ float bf2f(u16 h) {
    return __uint_as_float(((u32)h) << 16);
}
__device__ __forceinline__ float sigmoidf_(float x) {
    return 1.f / (1.f + __expf(-x));
}

// ---------------------------------------------------------------------------
// Tensor Data Mover: async 2D bf16 tile load global->LDS (D# packed by hand,
// see CDNA5 ISA). pad feature inserts 4 dwords after
// every 16 dwords so rows land on a 40-halfword LDS stride.
// ---------------------------------------------------------------------------
__device__ __forceinline__ void tdm_load_2d_bf16(u32 lds_off, const void* gptr,
                                                 u32 tile_d0, u32 tile_d1,
                                                 u64 row_stride_elems, int pad_en) {
#if USE_TDM
    u64 ga = (u64)gptr;
    v4u g0;
    g0[0] = 1u;                                   // count=1, user mode
    g0[1] = lds_off;                              // lds_addr (bytes)
    g0[2] = (u32)ga;                              // global_addr[31:0]
    g0[3] = (u32)(ga >> 32) | 0x80000000u;        // global_addr[56:32], type=2
    u32 w0 = (1u << 16);                          // data_size = 2B
    if (pad_en) w0 |= (1u << 20) | (3u << 22) | (3u << 25);  // pad every 16dw by 4dw
    const u32 td0 = 0x40000000u, td1 = 0x40000000u;  // generous tensor dims (no OOB clip)
    v8i g1;
    g1[0] = (int)w0;
    g1[1] = (int)((td0 & 0xFFFFu) << 16);                               // tensor_dim0 lo
    g1[2] = (int)(((td0 >> 16) & 0xFFFFu) | ((td1 & 0xFFFFu) << 16));   // dim0 hi | dim1 lo
    g1[3] = (int)(((td1 >> 16) & 0xFFFFu) | ((tile_d0 & 0xFFFFu) << 16)); // dim1 hi | tile_dim0
    g1[4] = (int)(tile_d1 & 0xFFFFu);                                   // tile_dim1 (tile_dim2=0)
    g1[5] = (int)(u32)(row_stride_elems & 0xFFFFFFFFull);               // dim0 stride lo32
    g1[6] = (int)(u32)((row_stride_elems >> 32) & 0xFFFFull);           // dim0 stride hi16
    g1[7] = 0;
    v4i z = {0, 0, 0, 0};
#if defined(__clang_major__) && __clang_major__ >= 23
    v8i z8 = {0, 0, 0, 0, 0, 0, 0, 0};
    __builtin_amdgcn_tensor_load_to_lds(g0, g1, z, z, z8, 0);
#else
    __builtin_amdgcn_tensor_load_to_lds(g0, g1, z, z, 0);
#endif
#endif
}

// ---------------------------------------------------------------------------
// Model dims
// ---------------------------------------------------------------------------
#define BB 64
#define SS 512
#define EE 256
#define HH 256
#define FourH 1024
#define GCDIM 256
#define FPAD 800   // fusion dim 774 padded to multiple of 32

// ---------------------------------------------------------------------------
// Generic f32 -> bf16 convert (grid-stride)
// ---------------------------------------------------------------------------
__global__ void cvt_bf_kernel(const float* __restrict__ in, u16* __restrict__ out, size_t n) {
    size_t i = (size_t)blockIdx.x * blockDim.x + threadIdx.x;
    size_t stride = (size_t)gridDim.x * blockDim.x;
    for (; i < n; i += stride) out[i] = f2bf(in[i]);
}

// fc1 weights [512, 774] -> bf16 padded [512, 800]
__global__ void cvt_fc1_kernel(const float* __restrict__ in, u16* __restrict__ out) {
    size_t i = (size_t)blockIdx.x * blockDim.x + threadIdx.x;
    size_t n = 512ull * FPAD;
    size_t stride = (size_t)gridDim.x * blockDim.x;
    for (; i < n; i += stride) {
        int row = (int)(i / FPAD), col = (int)(i % FPAD);
        out[i] = (col < 774) ? f2bf(in[(size_t)row * 774 + col]) : (u16)0;
    }
}

// ---------------------------------------------------------------------------
// Embedding gather
// ---------------------------------------------------------------------------
__global__ __launch_bounds__(256) void embed_kernel(const int* __restrict__ seq,
                                                    const float* __restrict__ table,
                                                    float* __restrict__ ef,
                                                    u16* __restrict__ ebf) {
    int row = blockIdx.x;            // b*S + s
    int e = threadIdx.x;
    int idx = seq[row];
    if ((unsigned)idx >= 50001u) idx = 0;
    float v = table[(size_t)idx * EE + e];
    ef[(size_t)row * EE + e] = v;
    ebf[(size_t)row * EE + e] = f2bf(v);
}

// ---------------------------------------------------------------------------
// WMMA GEMM:  C[M,N] = A[M,K](bf16,rm) * W[N,K](bf16,rm)^T + bias
// Tile 64x128, 256 threads (8 waves). TDM double-buffered LDS staging.
// ---------------------------------------------------------------------------
template <typename OutT>
__device__ __forceinline__ void store_out(OutT* p, float v);
template <> __device__ __forceinline__ void store_out<float>(float* p, float v) { *p = v; }
template <> __device__ __forceinline__ void store_out<u16>(u16* p, float v) { *p = f2bf(v); }

template <typename OutT>
__global__ __launch_bounds__(256) void wmma_gemm_bias(const u16* __restrict__ A,
                                                      const u16* __restrict__ W,
                                                      const float* __restrict__ bias,
                                                      OutT* __restrict__ C,
                                                      int M, int N, int K) {
    __shared__ u16 As[2][64][40];    // 40-half stride (80B rows): TDM pad feature
    __shared__ u16 Bs[2][128][40];

    const int m0 = blockIdx.x * 64;
    const int n0 = blockIdx.y * 128;
    const int tid = threadIdx.x;
    const int lane = tid & 31;
    const int wave = tid >> 5;
    const int msub = wave & 3;          // 16-row stripe
    const int nbase = (wave >> 2) * 64; // 64-col half

    const int nloc = lane & 15;
    const int kh = (lane >> 4) * 8;

    v8f acc[4] = {};
    const int nk = K >> 5;

#if USE_TDM
    if (tid < 32) {
        tdm_load_2d_bf16((u32)(size_t)&As[0][0][0], A + (size_t)m0 * K, 32, 64, (u64)K, 1);
        tdm_load_2d_bf16((u32)(size_t)&Bs[0][0][0], W + (size_t)n0 * K, 32, 128, (u64)K, 1);
    }
#else
    const int arow = tid >> 2, acol = (tid & 3) * 8;
    const int brow = tid >> 1, bcol = (tid & 1) * 16;
#endif

    int buf = 0;
    for (int kt = 0; kt < nk; ++kt) {
        const int k0 = kt * 32;
#if USE_TDM
        if (tid < 32) {
            if (kt + 1 < nk) {   // DMA next tiles while this one computes
                tdm_load_2d_bf16((u32)(size_t)&As[buf ^ 1][0][0],
                                 A + (size_t)m0 * K + (k0 + 32), 32, 64, (u64)K, 1);
                tdm_load_2d_bf16((u32)(size_t)&Bs[buf ^ 1][0][0],
                                 W + (size_t)n0 * K + (k0 + 32), 32, 128, (u64)K, 1);
                __builtin_amdgcn_s_wait_tensorcnt(2);   // current pair done, next in flight
            } else {
                __builtin_amdgcn_s_wait_tensorcnt(0);
            }
        }
        __syncthreads();
#else
        *(uint4*)&As[buf][arow][acol] = *(const uint4*)(A + (size_t)(m0 + arow) * K + k0 + acol);
        *(uint4*)&Bs[buf][brow][bcol] = *(const uint4*)(W + (size_t)(n0 + brow) * K + k0 + bcol);
        *(uint4*)&Bs[buf][brow][bcol + 8] = *(const uint4*)(W + (size_t)(n0 + brow) * K + k0 + bcol + 8);
        __syncthreads();
#endif

        V16 aU;
        const int mm = msub * 16 + nloc;
#pragma unroll
        for (int v = 0; v < 8; ++v) {
            int kb = ((v >> 2) << 4) + kh + ((v & 3) << 1);
            aU.u[v] = *(const u32*)&As[buf][mm][kb];
        }
#pragma unroll
        for (int j = 0; j < 4; ++j) {
            V16 bU;
            const int nn = nbase + j * 16 + nloc;
#pragma unroll
            for (int v = 0; v < 8; ++v) {
                int kb = ((v >> 2) << 4) + kh + ((v & 3) << 1);
                bU.u[v] = *(const u32*)&Bs[buf][nn][kb];
            }
            acc[j] = __builtin_amdgcn_wmma_f32_16x16x32_bf16(
                false, aU.v, false, bU.v, (short)0, acc[j], false, false);
        }
        __syncthreads();  // reads done before DMA may overwrite this buffer
        buf ^= 1;
    }

#pragma unroll
    for (int j = 0; j < 4; ++j) {
        int n = n0 + nbase + j * 16 + nloc;
        float bvv = bias ? bias[n] : 0.f;
#pragma unroll
        for (int r = 0; r < 8; ++r) {
            int m = m0 + msub * 16 + r + kh;
            store_out<OutT>(C + (size_t)m * N + n, acc[j][r] + bvv);
        }
    }
}

// ---------------------------------------------------------------------------
// Persistent bidirectional LSTM recurrence. grid.x = 2 (direction).
// 1024 threads = 32 waves cover the [64 x 1024] gate GEMM each step.
// Wave (msub, nn) owns gate subtiles {nn,nn+8} x {i,f,g,o} so the cell
// nonlinearity is per-lane in registers. h_t in LDS (bf16), c in VGPRs.
// Per-step x-gates slice [64,1024] is TDM-prefetched 1 step ahead into a
// double-buffered 2x128KB LDS region (DMA overlaps the WMMA K-loop).
// ---------------------------------------------------------------------------
#define HSTRIDE 264  // padded halves per h row

__global__ __launch_bounds__(1024) void lstm_recurrent(const u16* __restrict__ gates_bf, // [2,B,S,4H]
                                                       const u16* __restrict__ whh_bf,   // [2,4H,H]
                                                       float* __restrict__ out_f32) {    // [B,S,2H]
    const int dir = blockIdx.x;
    const u16* gx = gates_bf + (size_t)dir * BB * SS * FourH;
    const u32* whh32 = (const u32*)(whh_bf + (size_t)dir * FourH * HH);

    extern __shared__ u16 smem[];
    u16* hls = smem;                       // 64*HSTRIDE halves (33.8 KB)
    u16* gx0 = smem + 64 * HSTRIDE;        // 64*1024 halves (128 KB)
    u16* gx1 = gx0 + 64 * 1024;            // 64*1024 halves (128 KB)

    const int tid = threadIdx.x;
    for (int i = tid; i < 64 * HSTRIDE; i += 1024) hls[i] = 0;

    const int lane = tid & 31;
    const int wave = tid >> 5;
    const int msub = wave & 3;   // batch stripe
    const int nn = wave >> 2;    // 0..7 hidden-subtile group
    const int nloc = lane & 15;
    const int kh = (lane >> 4) * 8;

    float c_reg[16];
#pragma unroll
    for (int i = 0; i < 16; ++i) c_reg[i] = 0.f;

    // prefetch gates slice for t=0
    if (tid < 32) {
        const int s0 = dir ? (SS - 1) : 0;
        tdm_load_2d_bf16((u32)(size_t)gx0, gx + (size_t)s0 * FourH, FourH, 64,
                         (u64)SS * FourH, 0);
    }
    __syncthreads();

    int buf = 0;
    for (int t = 0; t < SS; ++t) {
        const int s = dir ? (SS - 1 - t) : t;

        // DMA next step's gate slice while we do this step's WMMAs
        if (tid < 32 && t + 1 < SS) {
            const int sn = dir ? (SS - 2 - t) : (t + 1);
            tdm_load_2d_bf16((u32)(size_t)(buf ? gx0 : gx1), gx + (size_t)sn * FourH,
                             FourH, 64, (u64)SS * FourH, 0);
        }

        v8f acc[2][4] = {};
        for (int kk = 0; kk < 8; ++kk) {
            const int k0 = kk * 32;
            V16 aU;
            const int mm = msub * 16 + nloc;
#pragma unroll
            for (int v = 0; v < 8; ++v) {
                int kb = k0 + ((v >> 2) << 4) + kh + ((v & 3) << 1);
                aU.u[v] = *(const u32*)&hls[mm * HSTRIDE + kb];
            }
#pragma unroll
            for (int q = 0; q < 2; ++q) {
#pragma unroll
                for (int g = 0; g < 4; ++g) {
                    const int n = g * 256 + (nn + 8 * q) * 16 + nloc;
                    V16 bU;
#pragma unroll
                    for (int v = 0; v < 8; ++v) {
                        int kb = k0 + ((v >> 2) << 4) + kh + ((v & 3) << 1);
                        bU.u[v] = whh32[(n * HH + kb) >> 1];  // WGP$ serves 4x reuse
                    }
                    acc[q][g] = __builtin_amdgcn_wmma_f32_16x16x32_bf16(
                        false, aU.v, false, bU.v, (short)0, acc[q][g], false, false);
                }
            }
        }

        if (tid < 32) {   // this step's gate DMA must have landed
            if (t + 1 < SS) __builtin_amdgcn_s_wait_tensorcnt(1);
            else            __builtin_amdgcn_s_wait_tensorcnt(0);
        }
        __syncthreads();  // + all reads of old h done before rewrites

        const u16* gxls = buf ? gx1 : gx0;
#pragma unroll
        for (int q = 0; q < 2; ++q) {
            const int j0 = (nn + 8 * q) * 16 + nloc;  // hidden column
#pragma unroll
            for (int r = 0; r < 8; ++r) {
                const int m = msub * 16 + r + kh;     // batch row
                const int gb = m * FourH;
                float gi = acc[q][0][r] + bf2f(gxls[gb + 0 * 256 + j0]);
                float gf = acc[q][1][r] + bf2f(gxls[gb + 1 * 256 + j0]);
                float gg = acc[q][2][r] + bf2f(gxls[gb + 2 * 256 + j0]);
                float go = acc[q][3][r] + bf2f(gxls[gb + 3 * 256 + j0]);
                float c = sigmoidf_(gf) * c_reg[q * 8 + r] + sigmoidf_(gi) * tanhf(gg);
                float h = sigmoidf_(go) * tanhf(c);
                c_reg[q * 8 + r] = c;
                hls[m * HSTRIDE + j0] = f2bf(h);
                out_f32[((size_t)m * SS + s) * 512 + dir * HH + j0] = h;
            }
        }
        __syncthreads();
        buf ^= 1;
    }
}

// ---------------------------------------------------------------------------
// GCN banded adjacency stencil -> bf16 (GEMM input)
// ---------------------------------------------------------------------------
__global__ __launch_bounds__(256) void banded_kernel(const float* __restrict__ g,
                                                     const float* __restrict__ mask,
                                                     u16* __restrict__ out) {
    int row = blockIdx.x;          // b*S + s
    int s = row & (SS - 1);
    int e = threadIdx.x;
    float m = mask[row];
    float mn = (s < SS - 1) ? mask[row + 1] : 0.f;
    float mp = (s > 0) ? mask[row - 1] : 0.f;
    float rs = m * (mp + mn) + 1e-8f;
    float cn = m * mn / rs, cp = m * mp / rs;
    float xn = (s < SS - 1) ? g[(size_t)(row + 1) * GCDIM + e] : 0.f;
    float xp = (s > 0) ? g[(size_t)(row - 1) * GCDIM + e] : 0.f;
    out[(size_t)row * GCDIM + e] = f2bf(cn * xn + cp * xp);
}

// residual + layernorm + relu, in place over y
__global__ __launch_bounds__(256) void ln_res_relu(float* __restrict__ y,
                                                   const float* __restrict__ xres,
                                                   const float* __restrict__ gamma,
                                                   const float* __restrict__ beta) {
    __shared__ float red[256];
    size_t row = blockIdx.x;
    int t = threadIdx.x;
    float v = y[row * GCDIM + t] + xres[row * GCDIM + t];
    red[t] = v;
    __syncthreads();
    for (int st = 128; st > 0; st >>= 1) { if (t < st) red[t] += red[t + st]; __syncthreads(); }
    float mu = red[0] * (1.f / GCDIM);
    __syncthreads();
    float d = v - mu;
    red[t] = d * d;
    __syncthreads();
    for (int st = 128; st > 0; st >>= 1) { if (t < st) red[t] += red[t + st]; __syncthreads(); }
    float var = red[0] * (1.f / GCDIM);
    float o = d * rsqrtf(var + 1e-5f) * gamma[t] + beta[t];
    y[row * GCDIM + t] = fmaxf(o, 0.f);
}

// ---------------------------------------------------------------------------
// Masked mean pooling over S: src [B,S,F] -> dst [B,F]
// ---------------------------------------------------------------------------
__global__ __launch_bounds__(256) void masked_mean(const float* __restrict__ src,
                                                   const float* __restrict__ mask,
                                                   float* __restrict__ dst, int F) {
    int b = blockIdx.x;
    float denom = 0.f;
    for (int s = 0; s < SS; ++s) denom += mask[b * SS + s];
    denom = fmaxf(denom, 1e-8f);
    for (int f = threadIdx.x; f < F; f += blockDim.x) {
        float acc = 0.f;
        for (int s = 0; s < SS; ++s)
            acc += src[((size_t)b * SS + s) * F + f] * mask[b * SS + s];
        dst[(size_t)b * F + f] = acc / denom;
    }
}

// Build fused bf16 [64, 800]: [lstm 512 | gcn 256 | aux 6 | zero pad 26]
__global__ void fuse_kernel(const float* __restrict__ lf, const float* __restrict__ gf,
                            const float* __restrict__ aux, u16* __restrict__ out) {
    int i = blockIdx.x * blockDim.x + threadIdx.x;
    if (i >= BB * FPAD) return;
    int b = i / FPAD, k = i % FPAD;
    float v;
    if (k < 512)      v = lf[b * 512 + k];
    else if (k < 768) v = gf[b * 256 + (k - 512)];
    else if (k < 774) v = aux[b * 6 + (k - 768)];
    else              v = 0.f;
    out[i] = f2bf(v);
}

__global__ void relu_cvt_kernel(const float* __restrict__ in, u16* __restrict__ out, int n) {
    int i = blockIdx.x * blockDim.x + threadIdx.x;
    if (i < n) out[i] = f2bf(fmaxf(in[i], 0.f));
}
__global__ void relu_kernel(float* __restrict__ x, int n) {
    int i = blockIdx.x * blockDim.x + threadIdx.x;
    if (i < n) x[i] = fmaxf(x[i], 0.f);
}

// heads: 640 outputs of head0 [64,10] then 320 of head1 [64,5]
__global__ void heads_kernel(const float* __restrict__ s,
                             const float* __restrict__ w0, const float* __restrict__ b0h,
                             const float* __restrict__ w1, const float* __restrict__ b1h,
                             float* __restrict__ out) {
    int gid = blockIdx.x * blockDim.x + threadIdx.x;
    if (gid >= 960) return;
    const float* w;
    float bias;
    int b;
    if (gid < 640) { b = gid / 10; int n = gid % 10; w = w0 + n * 256; bias = b0h[n]; }
    else { int r = gid - 640; b = r / 5; int n = r % 5; w = w1 + n * 256; bias = b1h[n]; }
    float acc = bias;
    const float* row = s + b * 256;
    for (int k = 0; k < 256; ++k) acc += row[k] * w[k];
    out[gid] = acc;
}

// ---------------------------------------------------------------------------
// Host orchestration
// ---------------------------------------------------------------------------
extern "C" void kernel_launch(void* const* d_in, const int* in_sizes, int n_in,
                              void* d_out, int out_size, void* d_ws, size_t ws_size,
                              hipStream_t stream) {
    const int*   seq      = (const int*)d_in[0];
    const float* mask     = (const float*)d_in[1];
    const float* aux      = (const float*)d_in[2];
    const float* emb_tab  = (const float*)d_in[3];
    const float* Wih0     = (const float*)d_in[4];   // [2,4H,E]
    const float* Whh0     = (const float*)d_in[5];   // [2,4H,H]
    const float* b0       = (const float*)d_in[6];   // [2,4H]
    const float* WihL     = (const float*)d_in[7];   // [4,2,4H,2H]
    const float* WhhL     = (const float*)d_in[8];   // [4,2,4H,H]
    const float* bL       = (const float*)d_in[9];   // [4,2,4H]
    const float* gcn_W    = (const float*)d_in[10];  // [3,GC,E]
    const float* gcn_b    = (const float*)d_in[11];
    const float* gcn_g    = (const float*)d_in[12];
    const float* gcn_be   = (const float*)d_in[13];
    const float* fc1_W    = (const float*)d_in[14];  // [512,774]
    const float* fc1_b    = (const float*)d_in[15];
    const float* fc2_W    = (const float*)d_in[16];  // [256,512]
    const float* fc2_b    = (const float*)d_in[17];
    const float* h0W      = (const float*)d_in[18];
    const float* h0b      = (const float*)d_in[19];
    const float* h1W      = (const float*)d_in[20];
    const float* h1b      = (const float*)d_in[21];

    char* w = (char*)d_ws;
    float* emb_f32 = (float*)(w + 0);                    // 33554432
    u16*   emb_bf  = (u16*)(w + 33554432ull);            // 16777216
    u16*   x_bf    = (u16*)(w + 50331648ull);            // 33554432
    float* out_f32 = (float*)(w + 83886080ull);          // 67108864
    u16*   gates   = (u16*)(w + 150994944ull);           // 134217728
    float* gA      = (float*)(w + 285212672ull);         // 33554432
    float* gB      = (float*)(w + 318767104ull);         // 33554432
    u16*   tmp_bf  = (u16*)(w + 352321536ull);           // 16777216
    u16*   wih_bf  = (u16*)(w + 369098752ull);           // 9437184
    u16*   whh_bf  = (u16*)(w + 378535936ull);           // 5242880
    u16*   gcnW_bf = (u16*)(w + 383778816ull);           // 393216
    u16*   fc1W_bf = (u16*)(w + 384172032ull);           // 819200
    u16*   fc2W_bf = (u16*)(w + 384991232ull);           // 262144
    float* lf      = (float*)(w + 385253376ull);         // 131072
    float* gf      = (float*)(w + 385384448ull);         // 65536
    u16*   fused   = (u16*)(w + 385449984ull);           // 102400
    float* s1f     = (float*)(w + 385552384ull);         // 131072
    u16*   s1bf    = (u16*)(w + 385683456ull);           // 65536
    float* s2f     = (float*)(w + 385748992ull);         // 65536
    float* outp    = (float*)d_out;

    // ---- weight conversion to bf16 ----
    cvt_bf_kernel<<<2048, 256, 0, stream>>>(Wih0, wih_bf, 2ull * FourH * EE);
    cvt_bf_kernel<<<2048, 256, 0, stream>>>(WihL, wih_bf + 2ull * FourH * EE, 8ull * FourH * 512);
    cvt_bf_kernel<<<2048, 256, 0, stream>>>(Whh0, whh_bf, 2ull * FourH * HH);
    cvt_bf_kernel<<<2048, 256, 0, stream>>>(WhhL, whh_bf + 2ull * FourH * HH, 8ull * FourH * HH);
    cvt_bf_kernel<<<512, 256, 0, stream>>>(gcn_W, gcnW_bf, 3ull * GCDIM * EE);
    cvt_fc1_kernel<<<512, 256, 0, stream>>>(fc1_W, fc1W_bf);
    cvt_bf_kernel<<<512, 256, 0, stream>>>(fc2_W, fc2W_bf, 256ull * 512);

    // ---- embedding ----
    embed_kernel<<<BB * SS, 256, 0, stream>>>(seq, emb_tab, emb_f32, emb_bf);

    const u32 lstm_lds = (64u * HSTRIDE + 2u * 64u * 1024u) * 2u;  // 295936 B

    // ---- 5-layer bidirectional LSTM ----
    for (int l = 0; l < 5; ++l) {
        const int K = (l == 0) ? EE : 512;
        const u16* Abf = (l == 0) ? emb_bf : x_bf;
        for (int d = 0; d < 2; ++d) {
            const u16* Wl = (l == 0)
                ? wih_bf + (size_t)d * FourH * EE
                : wih_bf + 2ull * FourH * EE + ((size_t)(l - 1) * 2 + d) * FourH * 512;
            const float* bptr = (l == 0) ? b0 + d * FourH : bL + ((size_t)(l - 1) * 2 + d) * FourH;
            wmma_gemm_bias<u16><<<dim3(BB * SS / 64, FourH / 128), 256, 0, stream>>>(
                Abf, Wl, bptr, gates + (size_t)d * BB * SS * FourH, BB * SS, FourH, K);
        }
        const u16* whh_l = (l == 0) ? whh_bf : whh_bf + 2ull * FourH * HH + (size_t)(l - 1) * 2 * FourH * HH;
        lstm_recurrent<<<2, 1024, lstm_lds, stream>>>(gates, whh_l, out_f32);
        if (l < 4)
            cvt_bf_kernel<<<4096, 256, 0, stream>>>(out_f32, x_bf, (size_t)BB * SS * 512);
    }

    // ---- GCN stack (ping-pong gA/gB, residual from previous activation) ----
    banded_kernel<<<BB * SS, 256, 0, stream>>>(emb_f32, mask, tmp_bf);
    wmma_gemm_bias<float><<<dim3(BB * SS / 64, GCDIM / 128), 256, 0, stream>>>(
        tmp_bf, gcnW_bf + 0ull * GCDIM * EE, gcn_b + 0 * GCDIM, gA, BB * SS, GCDIM, EE);
    ln_res_relu<<<BB * SS, 256, 0, stream>>>(gA, emb_f32, gcn_g + 0 * GCDIM, gcn_be + 0 * GCDIM);

    banded_kernel<<<BB * SS, 256, 0, stream>>>(gA, mask, tmp_bf);
    wmma_gemm_bias<float><<<dim3(BB * SS / 64, GCDIM / 128), 256, 0, stream>>>(
        tmp_bf, gcnW_bf + 1ull * GCDIM * EE, gcn_b + 1 * GCDIM, gB, BB * SS, GCDIM, EE);
    ln_res_relu<<<BB * SS, 256, 0, stream>>>(gB, gA, gcn_g + 1 * GCDIM, gcn_be + 1 * GCDIM);

    banded_kernel<<<BB * SS, 256, 0, stream>>>(gB, mask, tmp_bf);
    wmma_gemm_bias<float><<<dim3(BB * SS / 64, GCDIM / 128), 256, 0, stream>>>(
        tmp_bf, gcnW_bf + 2ull * GCDIM * EE, gcn_b + 2 * GCDIM, gA, BB * SS, GCDIM, EE);
    ln_res_relu<<<BB * SS, 256, 0, stream>>>(gA, gB, gcn_g + 2 * GCDIM, gcn_be + 2 * GCDIM);

    // ---- pooling + fusion ----
    masked_mean<<<BB, 256, 0, stream>>>(out_f32, mask, lf, 512);
    masked_mean<<<BB, 256, 0, stream>>>(gA, mask, gf, GCDIM);
    fuse_kernel<<<(BB * FPAD + 255) / 256, 256, 0, stream>>>(lf, gf, aux, fused);

    // ---- MLP head via WMMA ----
    wmma_gemm_bias<float><<<dim3(1, 512 / 128), 256, 0, stream>>>(fused, fc1W_bf, fc1_b, s1f, 64, 512, FPAD);
    relu_cvt_kernel<<<(64 * 512 + 255) / 256, 256, 0, stream>>>(s1f, s1bf, 64 * 512);
    wmma_gemm_bias<float><<<dim3(1, 256 / 128), 256, 0, stream>>>(s1bf, fc2W_bf, fc2_b, s2f, 64, 256, 512);
    relu_kernel<<<(64 * 256 + 255) / 256, 256, 0, stream>>>(s2f, 64 * 256);

    heads_kernel<<<4, 256, 0, stream>>>(s2f, h0W, h0b, h1W, h1b, outp);
}
